// STLClassifier_14903536517795
// MI455X (gfx1250) — matrix operand order repaired and checked by reading, not compiled, not run on hardware
//
#include <hip/hip_runtime.h>
#include <hip/hip_bf16.h>

typedef __attribute__((ext_vector_type(16))) __bf16 v16bf;
typedef __attribute__((ext_vector_type(8)))  float  v8f;

#define L_LEN    1024
#define F_DIM    16
#define D_DIM    16384      // L*F
#define H_DIM    128
#define C_DIM    10
#define PERIOD   24
#define PF       384        // PERIOD * F
#define KSEG     16         // split-K segments
#define KT_TOTAL 512        // 16384 / 32 k-tiles
#define KT_SEG   32         // k-tiles per segment (512/16)
#define NSTRIP   64         // 1024 / 16 m-strips

// ---------------------------------------------------------------------------
// K1a: centered moving average (window 25, zero-padded 'same', /25 always)
// ---------------------------------------------------------------------------
__global__ void __launch_bounds__(256) stl_trend_k(const float* __restrict__ x,
                                                   float* __restrict__ trend) {
    int tid = blockIdx.x * 256 + threadIdx.x;      // 0 .. 131071
    int f = tid & 15;
    int c = (tid >> 4) & 7;                        // 8 chunks of 128 t
    int b = tid >> 7;                              // 0 .. 1023
    const float* xs = x     + (long)b * D_DIM + f;
    float*       ts = trend + (long)b * D_DIM + f;
    int t0 = c * 128;
    float s = 0.f;
    int lo = t0 - 12; if (lo < 0) lo = 0;
    int hi = t0 + 12; if (hi > L_LEN - 1) hi = L_LEN - 1;
    for (int t = lo; t <= hi; ++t) s += xs[t * F_DIM];
    for (int t = t0; t < t0 + 128; ++t) {
        ts[t * F_DIM] = s * (1.0f / 25.0f);
        int add = t + 13, sub = t - 12;
        if (add <= L_LEN - 1) s += xs[add * F_DIM];
        if (sub >= 0)         s -= xs[sub * F_DIM];
    }
}

// ---------------------------------------------------------------------------
// K1b: seasonal means: Sm[b, p*16+f] = mean over t%24==p of (x - trend)
// ---------------------------------------------------------------------------
__global__ void __launch_bounds__(256) stl_seasonal_k(const float* __restrict__ x,
                                                      const float* __restrict__ trend,
                                                      float* __restrict__ Sm) {
    int tid = blockIdx.x * 256 + threadIdx.x;      // b*384 + p*16 + f
    int f = tid & 15;
    int p = (tid >> 4) % PERIOD;
    int b = tid / PF;
    const float* xs = x     + (long)b * D_DIM + f;
    const float* ts = trend + (long)b * D_DIM + f;
    float s = 0.f; int cnt = 0;
    for (int t = p; t < L_LEN; t += PERIOD) { s += xs[t * F_DIM] - ts[t * F_DIM]; ++cnt; }
    Sm[tid] = s / (float)cnt;
}

// ---------------------------------------------------------------------------
// K2: phase-folded weights: Wp[p*16+f, h] = sum_{t%24==p} We[(t*16+f), h]
// ---------------------------------------------------------------------------
__global__ void __launch_bounds__(128) stl_phase_w_k(const float* __restrict__ We1,
                                                     const float* __restrict__ We2,
                                                     float* __restrict__ Wp1,
                                                     float* __restrict__ Wp2) {
    int j = blockIdx.x;                            // p*16+f
    int h = threadIdx.x;
    int p = j >> 4, f = j & 15;
    float s1 = 0.f, s2 = 0.f;
    for (int t = p; t < L_LEN; t += PERIOD) {
        int r = (t * F_DIM + f) * H_DIM + h;
        s1 += We1[r];
        s2 += We2[r];
    }
    Wp1[j * H_DIM + h] = s1;
    Wp2[j * H_DIM + h] = s2;
}

// ---------------------------------------------------------------------------
// K2b: pack We0/We2 -> bf16 in WMMA B-fragment order.
// Layout: [kt (512)][nt (8)][lane (32)][16 bf16]; fragment elem 2v+h:
//   N = nt*16 + (lane&15), K = kt*32 + (v>=4?16:0) + (lane>=16?8:0) + 2*(v&3) + h
// ---------------------------------------------------------------------------
__global__ void __launch_bounds__(256) stl_pack_b_k(const float* __restrict__ We0,
                                                    const float* __restrict__ We2,
                                                    __bf16* __restrict__ Wb0,
                                                    __bf16* __restrict__ Wb2) {
    int tid = blockIdx.x * 256 + threadIdx.x;      // (kt*8+nt)*32+lane
    int lane = tid & 31;
    int nt   = (tid >> 5) & 7;
    int kt   = tid >> 8;
    int laneHi = lane >> 4, ml = lane & 15;
    int n = nt * 16 + ml;
    int kbase = kt * 32 + (laneHi << 3);
    v16bf f0, f2;
    #pragma unroll
    for (int e = 0; e < 16; ++e) {
        long idx = (long)(kbase + (e & 7) + ((e & 8) << 1)) * H_DIM + n;
        f0[e] = (__bf16)We0[idx];
        f2[e] = (__bf16)We2[idx];
    }
    ((v16bf*)Wb0)[tid] = f0;
    ((v16bf*)Wb2)[tid] = f2;
}

// ---------------------------------------------------------------------------
// K2c: pack trend / (x - trend) -> bf16 in WMMA A-fragment order.
// Layout: [strip (64)][kt (512)][lane (32)][16 bf16]; fragment elem 2v+h:
//   M = strip*16 + (lane&15)
//   K = kt*32 + (v>=4?16:0) + (lane>=16?8:0) + 2*(v&3) + h
// ---------------------------------------------------------------------------
__global__ void __launch_bounds__(256) stl_pack_a_k(const float* __restrict__ x,
                                                    const float* __restrict__ trend,
                                                    __bf16* __restrict__ Atr,
                                                    __bf16* __restrict__ Adt) {
    int tid = blockIdx.x * 256 + threadIdx.x;      // (strip*512+kt)*32+lane
    int lane  = tid & 31;
    int kt    = (tid >> 5) & 511;
    int strip = tid >> 14;
    int laneHi = lane >> 4, ml = lane & 15;
    long row = (long)(strip * 16 + ml) * D_DIM;
    v16bf aT, aD;
    #pragma unroll
    for (int v = 0; v < 8; ++v) {
        int k = kt * 32 + ((v & 4) << 2) + (laneHi << 3) + ((v & 3) << 1);
        float2 t2 = *(const float2*)(trend + row + k);
        float2 x2 = *(const float2*)(x     + row + k);
        aT[2 * v]     = (__bf16)t2.x;
        aT[2 * v + 1] = (__bf16)t2.y;
        aD[2 * v]     = (__bf16)(x2.x - t2.x);
        aD[2 * v + 1] = (__bf16)(x2.y - t2.y);
    }
    ((v16bf*)Atr)[tid] = aT;
    ((v16bf*)Adt)[tid] = aD;
}

// ---------------------------------------------------------------------------
// K3: big GEMM, bf16 WMMA, f32 accumulate, split-K, split-output, 2 m-strips
// per wave for B reuse.
//   z==0: enc0_part = trend @ We0 ; z==1: enc2_part = (x - trend) @ We2
// grid = (32 strip-pairs, 16 k-segments, 2 outputs); block = 1 wave.
// Per k-tile: 4 A-loads + 16 B-loads (b128) feeding 16 WMMAs (1.25 ld/wmma);
// each B fragment is consumed by two back-to-back WMMAs.
// ---------------------------------------------------------------------------
__global__ void __launch_bounds__(32) stl_gemm_wmma_k(const __bf16* __restrict__ Atr,
                                                      const __bf16* __restrict__ Adt,
                                                      const __bf16* __restrict__ Wb0,
                                                      const __bf16* __restrict__ Wb2,
                                                      float* __restrict__ partial) {
    const int lane   = threadIdx.x;
    const int laneHi = lane >> 4;
    const int ml     = lane & 15;
    const int sp     = blockIdx.x;        // strip pair: rows [sp*32, sp*32+32)
    const int seg    = blockIdx.y;
    const int z      = blockIdx.z;
    const int kt0    = seg * KT_SEG;

    const v16bf* A  = (const v16bf*)(z ? Adt : Atr);
    const v16bf* Bm = (const v16bf*)(z ? Wb2 : Wb0);
    const int ncol  = z ? 128 : 0;

    v8f zero;
    #pragma unroll
    for (int e = 0; e < 8; ++e) zero[e] = 0.f;
    v8f acc0[8], acc1[8];                 // strip 2*sp and 2*sp+1
    #pragma unroll
    for (int i = 0; i < 8; ++i) { acc0[i] = zero; acc1[i] = zero; }

    const long aBase0 = ((long)(2 * sp)     * KT_TOTAL) * 32 + lane;
    const long aBase1 = ((long)(2 * sp + 1) * KT_TOTAL) * 32 + lane;

    for (int kt = kt0; kt < kt0 + KT_SEG; ++kt) {
        v16bf a0 = A[aBase0 + (long)kt * 32];
        v16bf a1 = A[aBase1 + (long)kt * 32];
        #pragma unroll
        for (int nt = 0; nt < 8; ++nt) {
            v16bf b = Bm[((long)kt * 8 + nt) * 32 + lane];
            acc0[nt] = __builtin_amdgcn_wmma_f32_16x16x32_bf16(
                false, a0, false, b, (short)0, acc0[nt], false, false);
            acc1[nt] = __builtin_amdgcn_wmma_f32_16x16x32_bf16(
                false, a1, false, b, (short)0, acc1[nt], false, false);
        }
    }

    // C layout: lanes 0-15: M=r, N=lane; lanes 16-31: M=8+r, N=lane-16
    #pragma unroll
    for (int nt = 0; nt < 8; ++nt) {
        #pragma unroll
        for (int r = 0; r < 8; ++r) {
            int mrow = r + (laneHi << 3);
            int n    = nt * 16 + ml;
            long b0  = ((long)seg * 1024 + (2 * sp) * 16     + mrow) * 256;
            long b1  = ((long)seg * 1024 + (2 * sp + 1) * 16 + mrow) * 256;
            partial[b0 + ncol + n] = acc0[nt][r];
            partial[b1 + ncol + n] = acc1[nt][r];
        }
    }
}

// ---------------------------------------------------------------------------
// K4: split-K reduce + biases + tiny seasonal GEMMs + ReLU MLP + classifier
// ---------------------------------------------------------------------------
__global__ void __launch_bounds__(128) stl_head_k(const float* __restrict__ partial,
                                                  const float* __restrict__ Sm,
                                                  const float* __restrict__ Wp1,
                                                  const float* __restrict__ Wp2,
                                                  const float* __restrict__ be0,
                                                  const float* __restrict__ be1,
                                                  const float* __restrict__ be2,
                                                  const float* __restrict__ Wf1,
                                                  const float* __restrict__ bf1,
                                                  const float* __restrict__ Wf2,
                                                  const float* __restrict__ bf2,
                                                  float* __restrict__ out) {
    __shared__ float srow[PF];
    __shared__ float fused[3 * H_DIM];
    __shared__ float hbuf[H_DIM];
    const int b = blockIdx.x;
    const int h = threadIdx.x;

    for (int j = h; j < PF; j += 128) srow[j] = Sm[b * PF + j];
    __syncthreads();

    float e0 = be0[h], e2 = be2[h];
    for (int seg = 0; seg < KSEG; ++seg) {
        long base = ((long)seg * 1024 + b) * 256;
        e0 += partial[base + h];
        e2 += partial[base + 128 + h];
    }
    float e1 = be1[h];
    for (int j = 0; j < PF; ++j) {
        float sv = srow[j];
        e1 += sv * Wp1[j * H_DIM + h];
        e2 -= sv * Wp2[j * H_DIM + h];
    }
    fused[h]       = e0;
    fused[128 + h] = e1;
    fused[256 + h] = e2;
    __syncthreads();

    float hv = bf1[h];
    for (int j = 0; j < 3 * H_DIM; ++j) hv += fused[j] * Wf1[j * H_DIM + h];
    hbuf[h] = hv > 0.f ? hv : 0.f;
    __syncthreads();

    if (h < C_DIM) {
        float o = bf2[h];
        for (int j = 0; j < H_DIM; ++j) o += hbuf[j] * Wf2[j * C_DIM + h];
        out[b * C_DIM + h] = o;
    }
}

// ---------------------------------------------------------------------------
extern "C" void kernel_launch(void* const* d_in, const int* in_sizes, int n_in,
                              void* d_out, int out_size, void* d_ws, size_t ws_size,
                              hipStream_t stream) {
    const float* x   = (const float*)d_in[0];
    const float* We0 = (const float*)d_in[1];
    const float* be0 = (const float*)d_in[2];
    const float* We1 = (const float*)d_in[3];
    const float* be1 = (const float*)d_in[4];
    const float* We2 = (const float*)d_in[5];
    const float* be2 = (const float*)d_in[6];
    const float* Wf1 = (const float*)d_in[7];
    const float* bf1 = (const float*)d_in[8];
    const float* Wf2 = (const float*)d_in[9];
    const float* bf2 = (const float*)d_in[10];
    float* out = (float*)d_out;

    float* ws      = (float*)d_ws;
    float* trend   = ws;                          // 16777216 f
    float* Sm      = trend + 16777216;            //   393216 f
    float* Wp1     = Sm + 393216;                 //    49152 f
    float* Wp2     = Wp1 + 49152;                 //    49152 f
    float* partial = Wp2 + 49152;                 //  4194304 f
    __bf16* Wb0    = (__bf16*)(partial + 4194304);// 2097152 bf16
    __bf16* Wb2    = Wb0 + 2097152;               // 2097152 bf16
    __bf16* Atr    = Wb2 + 2097152;               // 16777216 bf16
    __bf16* Adt    = Atr + 16777216;              // 16777216 bf16

    stl_trend_k   <<<512, 256, 0, stream>>>(x, trend);
    stl_seasonal_k<<<1536, 256, 0, stream>>>(x, trend, Sm);
    stl_phase_w_k <<<PF, 128, 0, stream>>>(We1, We2, Wp1, Wp2);
    stl_pack_b_k  <<<512, 256, 0, stream>>>(We0, We2, Wb0, Wb2);
    stl_pack_a_k  <<<4096, 256, 0, stream>>>(x, trend, Atr, Adt);
    dim3 g3(NSTRIP / 2, KSEG, 2);
    stl_gemm_wmma_k<<<g3, 32, 0, stream>>>(Atr, Adt, Wb0, Wb2, partial);
    stl_head_k    <<<1024, 128, 0, stream>>>(partial, Sm, Wp1, Wp2,
                                             be0, be1, be2, Wf1, bf1, Wf2, bf2, out);
}
